// DecoderOnlyTransformer_17575006175506
// MI455X (gfx1250) — matrix operand order repaired and checked by reading, not compiled, not run on hardware
//
#include <hip/hip_runtime.h>
#include <hip/hip_bf16.h>

typedef _Float16 v4h  __attribute__((ext_vector_type(4)));
typedef _Float16 v8h  __attribute__((ext_vector_type(8)));
typedef _Float16 v16h __attribute__((ext_vector_type(16)));
typedef float    v8f  __attribute__((ext_vector_type(8)));

#define LSEQ   1024
#define DMODEL 1024
#define NHEAD  16
#define DHEAD  64
#define DFF    4096
#define NVOCAB 32000
#define NLAYER 12

// layout modes
constexpr int A_NORM  = 0;  // A[m*K + k]
constexpr int A_HEADK = 1;  // A[(k/64)*M*64 + m*64 + (k%64)]   (concat-head K gather)
constexpr int B_NORM  = 0;  // B[k*N + n]
constexpr int B_TRANS = 1;  // B[n*K + k]                        (K^T, emb^T)
constexpr int B_HEAD  = 2;  // B[(n/64)*K*64 + k*64 + (n%64)]    (W[H][D][64])
constexpr int C_NORM  = 0;  // C[m*N + n]
constexpr int C_HEAD  = 1;  // C[(n/64)*M*64 + m*64 + (n%64)]    ([H][L][64] output)
constexpr int EPI_NONE         = 0;
constexpr int EPI_BIAS         = 1;
constexpr int EPI_BIAS_RELU    = 2;
constexpr int EPI_SCALE_CAUSAL = 3;

#define M_BLK 64
#define N_BLK 128
#define K_BLK 64            // two 32-K WMMA sub-steps per staged tile
#define LDS_PAD 8

__device__ __forceinline__ v8h pack8(const float4 a, const float4 b) {
    v8h h;
    h[0] = (_Float16)a.x; h[1] = (_Float16)a.y; h[2] = (_Float16)a.z; h[3] = (_Float16)a.w;
    h[4] = (_Float16)b.x; h[5] = (_Float16)b.y; h[6] = (_Float16)b.z; h[7] = (_Float16)b.w;
    return h;
}

// -------------------- WMMA GEMM --------------------
// Requirements honored by every launch below: K % 64 == 0, M % 64 == 0.
// N may be ragged (e.g. 64 < N_BLK): loads are address-clamped, stores masked.
// 256 threads = 8 waves tiled 2(M) x 4(N); each wave owns 32x32 of C
// -> 2x2 subtiles x 2 K-substeps = 8 v_wmma_f32_16x16x32_f16 per staged tile.
template<int AM, int BM, int CM, int EPI>
__global__ __launch_bounds__(256)
void gemm_wmma(const float* __restrict__ A, const float* __restrict__ B,
               const float* __restrict__ bias, float* __restrict__ C,
               int M, int N, int K, float alpha,
               long sAz, long sBz, long sCz)
{
    __shared__ _Float16 As[2][M_BLK][K_BLK + LDS_PAD];   // row-major [m][k]
    __shared__ _Float16 Bs[2][N_BLK][K_BLK + LDS_PAD];   // transposed [n][k]

    const int tid  = threadIdx.x;
    const int lane = tid & 31;
    const int wid  = tid >> 5;
    const int wm   = (wid & 1) * 32;
    const int wn   = (wid >> 1) * 32;
    const int mBase = blockIdx.y * M_BLK;
    const int nBase = blockIdx.x * N_BLK;
    const int g   = lane >> 4;
    const int l16 = lane & 15;

    const float* Ab = A + (size_t)blockIdx.z * sAz;
    const float* Bb = B + (size_t)blockIdx.z * sBz;
    float*       Cb = C + (size_t)blockIdx.z * sCz;

    // ---- per-thread staging geometry (branch-free, vector loads) ----
    const int ar  = tid >> 2;            // A: row 0..63
    const int ac  = (tid & 3) * 16;      // A: 16 consecutive k  -> 4 x float4
    int gmA = mBase + ar; if (gmA > M - 1) gmA = M - 1;

    const int tnT  = tid >> 1;           // B_TRANS: n row 0..127
    const int tk0T = (tid & 1) * 32;     // 32 consecutive k     -> 8 x float4

    float4 fa[4];
    float4 fb[8];

    auto loadA = [&](int kb) {
#pragma unroll
        for (int c = 0; c < 4; ++c) {
            const int k0 = kb + ac + c * 4;
            size_t off;
            if (AM == A_NORM) off = (size_t)gmA * K + k0;
            else              off = ((size_t)(k0 >> 6) * M + gmA) * 64 + (k0 & 63);
            fa[c] = *(const float4*)(Ab + off);
        }
    };
    auto storeA = [&](int buf) {
        *(v8h*)&As[buf][ar][ac]     = pack8(fa[0], fa[1]);
        *(v8h*)&As[buf][ar][ac + 8] = pack8(fa[2], fa[3]);
    };

    auto loadB = [&](int kb) {
        if (BM == B_TRANS) {
            int gn = nBase + tnT; if (gn > N - 1) gn = N - 1;
            const float* s = Bb + (size_t)gn * K + kb + tk0T;
#pragma unroll
            for (int c = 0; c < 8; ++c) fb[c] = *(const float4*)(s + c * 4);
        } else {
            // two 4(k) x 4(n) micro-blocks per thread; lanes stride along n
            // so each row load is 32 lanes x 16B = 512B contiguous.
#pragma unroll
            for (int q = 0; q < 2; ++q) {
                const int qi = tid + q * 256;          // 0..511
                const int k0 = (qi >> 5) * 4;          // 0..60
                const int n0 = (qi & 31) * 4;          // 0..124
#pragma unroll
                for (int e = 0; e < 4; ++e) {
                    const int gk = kb + k0 + e;
                    if (BM == B_NORM) {
                        int gn = nBase + n0;
                        if (gn > N - 4) gn = N - 4;    // address clamp (N >= 64, mult of 4)
                        fb[q * 4 + e] = *(const float4*)(Bb + (size_t)gk * N + gn);
                    } else {                            // B_HEAD: N multiple of 128
                        const int gn = nBase + n0;
                        fb[q * 4 + e] = *(const float4*)(Bb + ((size_t)(gn >> 6) * K + gk) * 64 + (gn & 63));
                    }
                }
            }
        }
    };
    auto storeB = [&](int buf) {
        if (BM == B_TRANS) {
#pragma unroll
            for (int c = 0; c < 4; ++c)
                *(v8h*)&Bs[buf][tnT][tk0T + c * 8] = pack8(fb[2 * c], fb[2 * c + 1]);
        } else {
            // 4x4 register micro-transpose, one ds_store_b64 per output column
#pragma unroll
            for (int q = 0; q < 2; ++q) {
                const int qi = tid + q * 256;
                const int k0 = (qi >> 5) * 4;
                const int n0 = (qi & 31) * 4;
                float arr[4][4];
#pragma unroll
                for (int e = 0; e < 4; ++e) {
                    arr[e][0] = fb[q * 4 + e].x; arr[e][1] = fb[q * 4 + e].y;
                    arr[e][2] = fb[q * 4 + e].z; arr[e][3] = fb[q * 4 + e].w;
                }
#pragma unroll
                for (int j = 0; j < 4; ++j) {
                    v4h h;
                    h[0] = (_Float16)arr[0][j]; h[1] = (_Float16)arr[1][j];
                    h[2] = (_Float16)arr[2][j]; h[3] = (_Float16)arr[3][j];
                    *(v4h*)&Bs[buf][n0 + j][k0] = h;
                }
            }
        }
    };

    v8f acc[2][2];
#pragma unroll
    for (int mt = 0; mt < 2; ++mt)
#pragma unroll
        for (int nt = 0; nt < 2; ++nt)
#pragma unroll
            for (int r = 0; r < 8; ++r) acc[mt][nt][r] = 0.0f;

    loadA(0); loadB(0);
    storeA(0); storeB(0);
    __syncthreads();

    int buf = 0;
    for (int kb = 0; kb < K; kb += K_BLK) {
        const bool more = (kb + K_BLK) < K;
        if (more) { loadA(kb + K_BLK); loadB(kb + K_BLK); }   // overlap with WMMA below

#pragma unroll
        for (int ks = 0; ks < 2; ++ks) {
            const int ko = ks * 32;
            v16h av[2], bv[2];
#pragma unroll
            for (int mt = 0; mt < 2; ++mt) {
                const int row = wm + mt * 16 + l16;
                // A 16x32 frag: halves 0-15 -> K=g*16+h (contiguous in [m][k] tile)
                v8h a0 = *(const v8h*)&As[buf][row][ko + g * 16];
                v8h a1 = *(const v8h*)&As[buf][row][ko + g * 16 + 8];
#pragma unroll
                for (int i = 0; i < 8; ++i) { av[mt][i] = a0[i]; av[mt][i + 8] = a1[i]; }
            }
#pragma unroll
            for (int nt = 0; nt < 2; ++nt) {
                const int col = wn + nt * 16 + l16;
                // B 32x16 frag: halves 0-15 -> K=g*16+h (contiguous in [n][k] tile)
                v8h b0 = *(const v8h*)&Bs[buf][col][ko + g * 16];
                v8h b1 = *(const v8h*)&Bs[buf][col][ko + g * 16 + 8];
#pragma unroll
                for (int i = 0; i < 8; ++i) { bv[nt][i] = b0[i]; bv[nt][i + 8] = b1[i]; }
            }
#pragma unroll
            for (int mt = 0; mt < 2; ++mt)
#pragma unroll
                for (int nt = 0; nt < 2; ++nt)
                    acc[mt][nt] = __builtin_amdgcn_wmma_f32_16x16x32_f16(
                        false, av[mt], false, bv[nt], (short)0, acc[mt][nt], false, false);
        }

        if (more) { storeA(buf ^ 1); storeB(buf ^ 1); }       // other buffer: no hazard
        __syncthreads();
        buf ^= 1;
    }

    // ---- epilogue + store (C/D layout: N = lane%16, M = reg + 8*(lane/16)) ----
#pragma unroll
    for (int nt = 0; nt < 2; ++nt) {
        const int n = nBase + wn + nt * 16 + l16;
        const bool nok = n < N;
        float bs = 0.0f;
        if (EPI == EPI_BIAS || EPI == EPI_BIAS_RELU) bs = nok ? bias[n] : 0.0f;
#pragma unroll
        for (int mt = 0; mt < 2; ++mt) {
#pragma unroll
            for (int r = 0; r < 8; ++r) {
                const int m = mBase + wm + mt * 16 + r + 8 * g;
                if (nok && m < M) {
                    float v = acc[mt][nt][r];
                    if (EPI == EPI_SCALE_CAUSAL) {
                        v *= alpha;
                        if (n > m) v = -__builtin_inff();
                    } else if (EPI == EPI_BIAS) {
                        v += bs;
                    } else if (EPI == EPI_BIAS_RELU) {
                        v += bs;
                        v = v > 0.0f ? v : 0.0f;
                    }
                    if (CM == C_NORM) Cb[(size_t)m * N + n] = v;
                    else              Cb[((size_t)(n >> 6) * M + m) * 64 + (n & 63)] = v;
                }
            }
        }
    }
}

// -------------------- embedding + positional encoding --------------------
// a = 2 * emb[tok]*sqrt(3D) + PE   (reference: a = a + (a + pe))
__global__ __launch_bounds__(256)
void embed_pe(const int* __restrict__ tok, const float* __restrict__ emb,
              float* __restrict__ x)
{
    const int row = blockIdx.x;
    const int t   = tok[row];
    const float scale = sqrtf(3.0f * (float)DMODEL);
#pragma unroll
    for (int j = 0; j < 4; ++j) {
        const int d = threadIdx.x + j * 256;
        const float e = emb[(size_t)t * DMODEL + d] * scale;
        const int pair = d >> 1;
        const float base = __powf(10000.0f, -(float)(2 * pair) / (float)DMODEL);
        const float ang  = (float)row * base;
        const float pe   = (d & 1) ? __cosf(ang) : __sinf(ang);
        x[(size_t)row * DMODEL + d] = 2.0f * e + pe;
    }
}

// -------------------- row softmax (causal -inf already in place) --------------------
__global__ __launch_bounds__(256)
void softmax_rows(float* __restrict__ s)
{
    __shared__ float red[256];
    const int row = blockIdx.x;
    const int h   = blockIdx.y;
    float* p = s + ((size_t)h * LSEQ + row) * LSEQ;
    const int tid = threadIdx.x;

    float mx = -__builtin_inff();
    for (int i = tid; i < LSEQ; i += 256) mx = fmaxf(mx, p[i]);
    red[tid] = mx; __syncthreads();
    for (int st = 128; st > 0; st >>= 1) {
        if (tid < st) red[tid] = fmaxf(red[tid], red[tid + st]);
        __syncthreads();
    }
    mx = red[0]; __syncthreads();

    float sum = 0.0f;
    for (int i = tid; i < LSEQ; i += 256) {
        const float e = __expf(p[i] - mx);   // exp(-inf) = 0 handles the mask
        p[i] = e; sum += e;
    }
    red[tid] = sum; __syncthreads();
    for (int st = 128; st > 0; st >>= 1) {
        if (tid < st) red[tid] += red[tid + st];
        __syncthreads();
    }
    const float inv = 1.0f / red[0];
    for (int i = tid; i < LSEQ; i += 256) p[i] *= inv;
}

// -------------------- residual add + LayerNorm --------------------
// x = LN(x + r) * s + b   (in place over x), D = 1024, 256 threads/row
__global__ __launch_bounds__(256)
void ln_residual(float* __restrict__ x, const float* __restrict__ r,
                 const float* __restrict__ s, const float* __restrict__ b)
{
    __shared__ float red[256];
    const int row = blockIdx.x;
    float* xp = x + (size_t)row * DMODEL;
    const float* rp = r + (size_t)row * DMODEL;
    const int tid = threadIdx.x;

    float v[4];
    float sum = 0.0f;
#pragma unroll
    for (int j = 0; j < 4; ++j) {
        const int d = tid + j * 256;
        v[j] = xp[d] + rp[d];
        sum += v[j];
    }
    red[tid] = sum; __syncthreads();
    for (int st = 128; st > 0; st >>= 1) {
        if (tid < st) red[tid] += red[tid + st];
        __syncthreads();
    }
    const float mu = red[0] * (1.0f / DMODEL);
    __syncthreads();

    float sq = 0.0f;
#pragma unroll
    for (int j = 0; j < 4; ++j) { const float d = v[j] - mu; sq += d * d; }
    red[tid] = sq; __syncthreads();
    for (int st = 128; st > 0; st >>= 1) {
        if (tid < st) red[tid] += red[tid + st];
        __syncthreads();
    }
    const float var  = red[0] * (1.0f / DMODEL);
    const float rstd = rsqrtf(var + 1e-5f);
#pragma unroll
    for (int j = 0; j < 4; ++j) {
        const int d = tid + j * 256;
        xp[d] = (v[j] - mu) * rstd * s[d] + b[d];
    }
}

// -------------------- driver --------------------
extern "C" void kernel_launch(void* const* d_in, const int* in_sizes, int n_in,
                              void* d_out, int out_size, void* d_ws, size_t ws_size,
                              hipStream_t stream)
{
    (void)in_sizes; (void)n_in; (void)out_size; (void)ws_size;

    const int*   tokens = (const int*)  d_in[0];
    const float* emb    = (const float*)d_in[1];
    const float* finalb = (const float*)d_in[2];
    const float* Wq     = (const float*)d_in[3];
    const float* Wk     = (const float*)d_in[4];
    const float* Wv     = (const float*)d_in[5];
    const float* Wo     = (const float*)d_in[6];
    const float* ln1s   = (const float*)d_in[7];
    const float* ln1b   = (const float*)d_in[8];
    const float* W1     = (const float*)d_in[9];
    const float* b1     = (const float*)d_in[10];
    const float* W2     = (const float*)d_in[11];
    const float* b2     = (const float*)d_in[12];
    const float* ln2s   = (const float*)d_in[13];
    const float* ln2b   = (const float*)d_in[14];

    const size_t LD = (size_t)LSEQ * DMODEL;          // 1M floats
    float* x      = (float*)d_ws;                     // [L,D]
    float* t0     = x + 1 * LD;                       // [L,D] gemm tmp
    float* q      = x + 2 * LD;                       // [H,L,64]
    float* kk     = x + 3 * LD;                       // [H,L,64]
    float* vv     = x + 4 * LD;                       // [H,L,64]
    float* o      = x + 5 * LD;                       // [H,L,64]
    float* scores = x + 6 * LD;                       // [H,L,L] = 16*LD
    float* ffn    = x + 22 * LD;                      // [L,FF]  = 4*LD

    const dim3 blk(256);
    const float attn_scale = 0.03125f;                // 1/sqrt(D=1024)

    embed_pe<<<LSEQ, blk, 0, stream>>>(tokens, emb, x);

    for (int i = 0; i < NLAYER; ++i) {
        const size_t wOff = (size_t)i * NHEAD * DMODEL * DHEAD;
        const size_t dOff = (size_t)i * DMODEL;
        const size_t fOff = (size_t)i * DFF;

        const dim3 gqkv((NHEAD * DHEAD + N_BLK - 1) / N_BLK, (LSEQ + M_BLK - 1) / M_BLK, 1);
        gemm_wmma<A_NORM, B_HEAD, C_HEAD, EPI_NONE><<<gqkv, blk, 0, stream>>>(
            x, Wq + wOff, nullptr, q, LSEQ, NHEAD * DHEAD, DMODEL, 1.0f, 0, 0, 0);
        gemm_wmma<A_NORM, B_HEAD, C_HEAD, EPI_NONE><<<gqkv, blk, 0, stream>>>(
            x, Wk + wOff, nullptr, kk, LSEQ, NHEAD * DHEAD, DMODEL, 1.0f, 0, 0, 0);
        gemm_wmma<A_NORM, B_HEAD, C_HEAD, EPI_NONE><<<gqkv, blk, 0, stream>>>(
            x, Wv + wOff, nullptr, vv, LSEQ, NHEAD * DHEAD, DMODEL, 1.0f, 0, 0, 0);

        // scores[h] = (Q_h K_h^T) * scale with causal -inf
        const dim3 gsc((LSEQ + N_BLK - 1) / N_BLK, (LSEQ + M_BLK - 1) / M_BLK, NHEAD);
        gemm_wmma<A_NORM, B_TRANS, C_NORM, EPI_SCALE_CAUSAL><<<gsc, blk, 0, stream>>>(
            q, kk, nullptr, scores, LSEQ, LSEQ, DHEAD, attn_scale,
            (long)(LSEQ * DHEAD), (long)(LSEQ * DHEAD), (long)LSEQ * LSEQ);

        const dim3 gsm(LSEQ, NHEAD);
        softmax_rows<<<gsm, blk, 0, stream>>>(scores);

        // o[h] = P_h @ V_h
        const dim3 gpv((DHEAD + N_BLK - 1) / N_BLK, (LSEQ + M_BLK - 1) / M_BLK, NHEAD);
        gemm_wmma<A_NORM, B_NORM, C_NORM, EPI_NONE><<<gpv, blk, 0, stream>>>(
            scores, vv, nullptr, o, LSEQ, DHEAD, LSEQ, 1.0f,
            (long)LSEQ * LSEQ, (long)(LSEQ * DHEAD), (long)(LSEQ * DHEAD));

        // t0 = concat(o) @ Wo   (A gathered across head blocks)
        const dim3 gwo((DMODEL + N_BLK - 1) / N_BLK, (LSEQ + M_BLK - 1) / M_BLK, 1);
        gemm_wmma<A_HEADK, B_NORM, C_NORM, EPI_NONE><<<gwo, blk, 0, stream>>>(
            o, Wo + wOff, nullptr, t0, LSEQ, DMODEL, NHEAD * DHEAD, 1.0f, 0, 0, 0);

        ln_residual<<<LSEQ, blk, 0, stream>>>(x, t0, ln1s + dOff, ln1b + dOff);

        const dim3 gf1((DFF + N_BLK - 1) / N_BLK, (LSEQ + M_BLK - 1) / M_BLK, 1);
        gemm_wmma<A_NORM, B_NORM, C_NORM, EPI_BIAS_RELU><<<gf1, blk, 0, stream>>>(
            x, W1 + (size_t)i * DMODEL * DFF, b1 + fOff, ffn, LSEQ, DFF, DMODEL, 1.0f, 0, 0, 0);

        const dim3 gf2((DMODEL + N_BLK - 1) / N_BLK, (LSEQ + M_BLK - 1) / M_BLK, 1);
        gemm_wmma<A_NORM, B_NORM, C_NORM, EPI_BIAS><<<gf2, blk, 0, stream>>>(
            ffn, W2 + (size_t)i * DFF * DMODEL, b2 + dOff, t0, LSEQ, DMODEL, DFF, 1.0f, 0, 0, 0);

        ln_residual<<<LSEQ, blk, 0, stream>>>(x, t0, ln2s + dOff, ln2b + dOff);
    }

    // logits = x @ emb^T + final_b  -> d_out
    const dim3 gl((NVOCAB + N_BLK - 1) / N_BLK, (LSEQ + M_BLK - 1) / M_BLK, 1);
    gemm_wmma<A_NORM, B_TRANS, C_NORM, EPI_BIAS><<<gl, blk, 0, stream>>>(
        x, emb, finalb, (float*)d_out, LSEQ, NVOCAB, DMODEL, 1.0f, 0, 0, 0);
}
